// ExchangeBlock_83236466196511
// MI455X (gfx1250) — matrix operand order repaired and checked by reading, not compiled, not run on hardware
//
#include <hip/hip_runtime.h>
#include <hip/hip_bf16.h>

typedef __attribute__((ext_vector_type(16))) __bf16 v16bf;
typedef __attribute__((ext_vector_type(8)))  float  v8f;

#define NL0   32
#define NL1   16
#define NL2   8
#define NS    512
#define NBB   256
#define NEDGE 131072
#define NFEAT 120
#define KZ    1344
#define H1    1024

__device__ __forceinline__ __bf16 f2bf(float f) {
    union { float f; unsigned u; } x; x.f = f;
    unsigned r = x.u + 0x7fffu + ((x.u >> 16) & 1u);   // round-to-nearest-even
    unsigned short h = (unsigned short)(r >> 16);
    return __builtin_bit_cast(__bf16, h);
}
__device__ __forceinline__ float bf2f(__bf16 b) {
    unsigned short h = __builtin_bit_cast(unsigned short, b);
    union { unsigned u; float f; } x; x.u = ((unsigned)h) << 16;
    return x.f;
}

// ---------------------------------------------------------------------------
// bf16 WMMA GEMM:  C[M,N] = A[M,K] * Bt[N,K]^T (+bias)(+silu)
// Block tile 128x128, 8 waves (4M x 2N), wave tile 32x64 (8 accumulators),
// K step 32, ping-pong LDS double buffering.
// ---------------------------------------------------------------------------
template<int OUT_BF16, int SILU, int HAS_BIAS>
__global__ __launch_bounds__(256)
void gemm_bf16_kernel(const __bf16* __restrict__ A,
                      const __bf16* __restrict__ Bt,
                      const float*  __restrict__ bias,
                      void* __restrict__ Cout,
                      int M, int Nd, int K)
{
    __shared__ __align__(16) __bf16 sA[2][128 * 32];
    __shared__ __align__(16) __bf16 sB[2][128 * 32];

    const int tid  = threadIdx.x;
    const int lane = tid & 31;
    const int wave = tid >> 5;
    const int wm   = wave & 3;      // 0..3  (M)
    const int wn   = wave >> 2;     // 0..1  (N)
    const int m0   = blockIdx.x * 128;
    const int n0   = blockIdx.y * 128;
    const int half = lane >> 4;
    const int rr   = lane & 15;

    // cooperative tile-load addressing: 128 rows x 32 bf16 (64B/row), 2 threads/row
    const int lrow = tid >> 1;            // 0..127
    const int lcol = (tid & 1) * 16;      // bf16 column 0 or 16
    const __bf16* gA = A  + (size_t)(m0 + lrow) * K + lcol;
    const __bf16* gB = Bt + (size_t)(n0 + lrow) * K + lcol;
    __bf16* qA = &sA[0][lrow * 32 + lcol];
    __bf16* qB = &sB[0][lrow * 32 + lcol];

    auto loadTiles = [&](int k0, int buf) {
        uint4 a0 = *(const uint4*)(gA + k0);
        uint4 a1 = *(const uint4*)(gA + k0 + 8);
        uint4 b0 = *(const uint4*)(gB + k0);
        uint4 b1 = *(const uint4*)(gB + k0 + 8);
        __bf16* da = qA + buf * (128 * 32);
        __bf16* db = qB + buf * (128 * 32);
        *(uint4*)(da)     = a0;
        *(uint4*)(da + 8) = a1;
        *(uint4*)(db)     = b0;
        *(uint4*)(db + 8) = b1;
    };

    v8f acc[2][4] = {};
    const int KT = K >> 5;

    loadTiles(0, 0);
    __syncthreads();

    for (int kt = 0; kt < KT; ++kt) {
        const int cur = kt & 1;
        if (kt + 1 < KT) loadTiles((kt + 1) << 5, cur ^ 1);   // overlap with WMMA below

        const __bf16* a_base = &sA[cur][0];
        const __bf16* b_base = &sB[cur][0];

        // B fragments (32x16 each; contiguous k order: K = half*16..+15 per lane half)
        union { uint4 u[2]; v16bf v; } fb[4];
        for (int j = 0; j < 4; ++j) {
            const __bf16* pb = b_base + (wn * 64 + j * 16 + rr) * 32;
            fb[j].u[0] = *(const uint4*)(pb + half * 16);
            fb[j].u[1] = *(const uint4*)(pb + half * 16 + 8);
        }
        // A fragments (16x32 each; same contiguous k order as B), reused across 4 WMMAs
        for (int i = 0; i < 2; ++i) {
            union { uint4 u[2]; v16bf v; } fa;
            const __bf16* pa = a_base + (wm * 32 + i * 16 + rr) * 32;
            fa.u[0] = *(const uint4*)(pa + half * 16);
            fa.u[1] = *(const uint4*)(pa + half * 16 + 8);
            for (int j = 0; j < 4; ++j)
                acc[i][j] = __builtin_amdgcn_wmma_f32_16x16x32_bf16(
                    false, fa.v, false, fb[j].v, (short)0, acc[i][j], false, false);
        }
        __syncthreads();
    }

    // epilogue: C/D layout -> element q is row (half*8 + q), col = lane&15
    const int mbase = m0 + wm * 32 + half * 8;
    for (int i = 0; i < 2; ++i) {
        for (int j = 0; j < 4; ++j) {
            const int n = n0 + wn * 64 + j * 16 + rr;
            const float bv = HAS_BIAS ? bias[n] : 0.0f;
            for (int q = 0; q < 8; ++q) {
                float v = acc[i][j][q] + bv;
                if (SILU) v = v / (1.0f + __expf(-v));
                const size_t off = (size_t)(mbase + i * 16 + q) * Nd + n;
                if (OUT_BF16) ((__bf16*)Cout)[off] = f2bf(v);
                else          ((float*)Cout)[off]  = v;
            }
        }
    }
}

// ---------------------------------------------------------------------------
// Weight prep: f32 [K][N] -> bf16 [N][K]
// ---------------------------------------------------------------------------
__global__ __launch_bounds__(256)
void transpose_to_bf16(const float* __restrict__ W, __bf16* __restrict__ Wt,
                       int K, int Nd)
{
    int idx = blockIdx.x * 256 + threadIdx.x;
    if (idx >= K * Nd) return;
    int n = idx % Nd, k = idx / Nd;
    Wt[(size_t)n * K + k] = f2bf(W[idx]);
}

// Build symmetric combined irrep weight:  Wcat_t [NS][KZ] (bf16), scaling folded in.
__global__ __launch_bounds__(256)
void build_wcat_kernel(const float* __restrict__ W0, const float* __restrict__ W1,
                       const float* __restrict__ W2, __bf16* __restrict__ Wt)
{
    int idx = blockIdx.x * 256 + threadIdx.x;     // over NS*KZ
    if (idx >= NS * KZ) return;
    int k = idx % KZ, w = idx / KZ;
    const float fan = sqrtf((float)(NL0 * NL0 + NL1 * NL1 + NL2 * NL2)); // sqrt(1344)
    float val;
    if (k < 1024) {
        int u = k >> 5, v = k & 31;
        val = 0.5f * (W0[(u * 32 + v) * NS + w] + W0[(v * 32 + u) * NS + w]) / fan;
    } else if (k < 1280) {
        int t = k - 1024, u = t >> 4, v = t & 15;
        val = 0.5f * (W1[(u * 16 + v) * NS + w] + W1[(v * 16 + u) * NS + w]) / (fan * sqrtf(3.0f));
    } else {
        int t = k - 1280, u = t >> 3, v = t & 7;
        val = 0.5f * (W2[(u * 8 + v) * NS + w] + W2[(v * 8 + u) * NS + w]) / (fan * sqrtf(5.0f));
    }
    Wt[(size_t)w * KZ + k] = f2bf(val);
}

// ---------------------------------------------------------------------------
// Edge prep: geometry -> r, sin-embedding (bf16), gathered pair features z (bf16)
// One wave32 per edge, 8 edges per block.
// ---------------------------------------------------------------------------
__global__ __launch_bounds__(256)
void edge_prep_kernel(const float* __restrict__ nodes, const float* __restrict__ pos,
                      const float* __restrict__ cell,  const float* __restrict__ edge_shift,
                      const int*   __restrict__ edge_index, const int* __restrict__ batch_vec,
                      __bf16* __restrict__ emb_out, __bf16* __restrict__ z_out,
                      int e0, int CE)
{
    __shared__ float sfeat[8][2][NFEAT];
    __shared__ float sr[8];
    const int wave = threadIdx.x >> 5, lane = threadIdx.x & 31;
    const int ce = blockIdx.x * 8 + wave;
    const int e  = e0 + ce;
    const int src = edge_index[e];
    const int dst = edge_index[NEDGE + e];

    for (int i = lane; i < NFEAT; i += 32) {
        sfeat[wave][0][i] = nodes[(size_t)src * NFEAT + i];
        sfeat[wave][1][i] = nodes[(size_t)dst * NFEAT + i];
    }
    if (lane == 0) {
        const float* C = cell + batch_vec[src] * 9;
        float s0 = edge_shift[e * 3 + 0], s1 = edge_shift[e * 3 + 1], s2 = edge_shift[e * 3 + 2];
        float rx = pos[dst * 3 + 0] - pos[src * 3 + 0] + s0 * C[0] + s1 * C[3] + s2 * C[6];
        float ry = pos[dst * 3 + 1] - pos[src * 3 + 1] + s0 * C[1] + s1 * C[4] + s2 * C[7];
        float rz = pos[dst * 3 + 2] - pos[src * 3 + 2] + s0 * C[2] + s1 * C[5] + s2 * C[8];
        float dist = sqrtf(rx * rx + ry * ry + rz * rz);
        sr[wave] = 1.0f / (dist + 1e-6f);
    }
    __syncthreads();

    const float r    = sr[wave];
    const float invr = 1.0f / r;                       // = dist + 1e-6
    const float cpre = sqrtf(2.0f / 7.0f);
    const float warg = 3.14159265358979f * r / 7.0f;
    for (int j = 0; j < NBB / 32; ++j) {
        int n = j * 32 + lane;
        emb_out[(size_t)ce * NBB + n] = f2bf(cpre * __sinf((float)(n + 1) * warg) * invr);
    }

    const float* xs = sfeat[wave][0];
    const float* xd = sfeat[wave][1];
    __bf16* zrow = z_out + (size_t)ce * KZ;
    for (int j = 0; j < KZ / 32; ++j) {
        int k = j * 32 + lane;
        float z;
        if (k < 1024) {
            z = xs[k >> 5] * xd[k & 31];
        } else if (k < 1280) {
            int t = k - 1024, u = t >> 4, v = t & 15;
            const float* a = xs + 32 + u * 3; const float* b = xd + 32 + v * 3;
            z = a[0] * b[0] + a[1] * b[1] + a[2] * b[2];
        } else {
            int t = k - 1280, u = t >> 3, v = t & 7;
            const float* a = xs + 80 + u * 5; const float* b = xd + 80 + v * 5;
            z = a[0] * b[0] + a[1] * b[1] + a[2] * b[2] + a[3] * b[3] + a[4] * b[4];
        }
        zrow[k] = f2bf(z);
    }
}

// ---------------------------------------------------------------------------
// LayerNorm(mixed)*ln_g+ln_b, then * distfilter -> reg (bf16). One wave per edge.
// ---------------------------------------------------------------------------
__global__ __launch_bounds__(256)
void ln_filter_kernel(const float* __restrict__ mixed, const float* __restrict__ filt,
                      const float* __restrict__ ln_g, const float* __restrict__ ln_b,
                      __bf16* __restrict__ reg, int CE)
{
    const int wave = threadIdx.x >> 5, lane = threadIdx.x & 31;
    const int ce = blockIdx.x * 8 + wave;
    const float* row = mixed + (size_t)ce * NS;

    float vals[16];
    float s = 0.0f;
    for (int j = 0; j < 16; ++j) { vals[j] = row[j * 32 + lane]; s += vals[j]; }
    for (int o = 16; o >= 1; o >>= 1) s += __shfl_xor(s, o);
    const float mu = s * (1.0f / NS);
    float v = 0.0f;
    for (int j = 0; j < 16; ++j) { float d = vals[j] - mu; v += d * d; }
    for (int o = 16; o >= 1; o >>= 1) v += __shfl_xor(v, o);
    const float rstd = rsqrtf(v * (1.0f / NS) + 1e-5f);
    for (int j = 0; j < 16; ++j) {
        int n = j * 32 + lane;
        float y = (vals[j] - mu) * rstd * ln_g[n] + ln_b[n];
        reg[(size_t)ce * NS + n] = f2bf(y * filt[(size_t)ce * NS + n]);
    }
}

// ---------------------------------------------------------------------------
// Final projection: out[e] = h2[e,:]·mo_w + mo_b. One wave per edge.
// ---------------------------------------------------------------------------
__global__ __launch_bounds__(256)
void out_kernel(const __bf16* __restrict__ h2, const float* __restrict__ mo_w,
                const float* __restrict__ mo_b, float* __restrict__ out, int e0)
{
    const int wave = threadIdx.x >> 5, lane = threadIdx.x & 31;
    const int ce = blockIdx.x * 8 + wave;
    const __bf16* row = h2 + (size_t)ce * H1;
    float s = 0.0f;
    for (int j = 0; j < H1 / 32; ++j) {
        int n = j * 32 + lane;
        s += bf2f(row[n]) * mo_w[n];
    }
    for (int o = 16; o >= 1; o >>= 1) s += __shfl_xor(s, o);
    if (lane == 0) out[e0 + ce] = s + mo_b[0];
}

// ---------------------------------------------------------------------------
extern "C" void kernel_launch(void* const* d_in, const int* in_sizes, int n_in,
                              void* d_out, int out_size, void* d_ws, size_t ws_size,
                              hipStream_t stream)
{
    (void)in_sizes; (void)n_in; (void)out_size;
    const float* nodes      = (const float*)d_in[0];
    const float* pos        = (const float*)d_in[1];
    const float* cell       = (const float*)d_in[2];
    const float* edge_shift = (const float*)d_in[3];
    const float* W0         = (const float*)d_in[4];
    const float* W1         = (const float*)d_in[5];
    const float* W2         = (const float*)d_in[6];
    const float* ln_g       = (const float*)d_in[7];
    const float* ln_b       = (const float*)d_in[8];
    const float* df_w1      = (const float*)d_in[9];
    const float* df_b1      = (const float*)d_in[10];
    const float* df_w2      = (const float*)d_in[11];
    const float* df_b2      = (const float*)d_in[12];
    const float* mi_w1      = (const float*)d_in[13];
    const float* mi_b1      = (const float*)d_in[14];
    const float* mi_w2      = (const float*)d_in[15];
    const float* mi_b2      = (const float*)d_in[16];
    const float* mo_w       = (const float*)d_in[17];
    const float* mo_b       = (const float*)d_in[18];
    const int*   edge_index = (const int*)d_in[19];
    const int*   batch_vec  = (const int*)d_in[20];
    float* out = (float*)d_out;

    char* p = (char*)d_ws;
    auto alloc = [&](size_t bytes) -> char* {
        char* q = p; p += (bytes + 255) & ~(size_t)255; return q;
    };

    // bf16 weights, transposed to [N][K]
    __bf16* w_df1 = (__bf16*)alloc((size_t)1024 * 256 * 2);
    __bf16* w_df2 = (__bf16*)alloc((size_t)512 * 1024 * 2);
    __bf16* w_cat = (__bf16*)alloc((size_t)NS * KZ * 2);
    __bf16* w_mi1 = (__bf16*)alloc((size_t)1024 * 512 * 2);
    __bf16* w_mi2 = (__bf16*)alloc((size_t)1024 * 1024 * 2);
    const size_t fixed = (size_t)(p - (char*)d_ws);

    // chunk sizing: per-edge activation bytes; cap at 8192 edges (~100MB) to stay L2-resident
    const size_t per_edge = (size_t)(NBB * 2 + KZ * 2 + H1 * 2 + H1 * 2 + NS * 4 + NS * 4 + NS * 2) + 32;
    int CE = 8192;
    while (CE > 128 && fixed + (size_t)CE * per_edge > ws_size) CE >>= 1;

    __bf16* ws_emb  = (__bf16*)alloc((size_t)CE * NBB * 2);
    __bf16* ws_z    = (__bf16*)alloc((size_t)CE * KZ * 2);
    __bf16* ws_h    = (__bf16*)alloc((size_t)CE * H1 * 2);
    __bf16* ws_h2   = (__bf16*)alloc((size_t)CE * H1 * 2);
    float*  ws_filt = (float*) alloc((size_t)CE * NS * 4);
    float*  ws_mix  = (float*) alloc((size_t)CE * NS * 4);
    __bf16* ws_reg  = (__bf16*)alloc((size_t)CE * NS * 2);

    // ---- weight prep (cheap, every call: deterministic) ----
    transpose_to_bf16<<<(256 * 1024 + 255) / 256, 256, 0, stream>>>(df_w1, w_df1, 256, 1024);
    transpose_to_bf16<<<(1024 * 512 + 255) / 256, 256, 0, stream>>>(df_w2, w_df2, 1024, 512);
    transpose_to_bf16<<<(512 * 1024 + 255) / 256, 256, 0, stream>>>(mi_w1, w_mi1, 512, 1024);
    transpose_to_bf16<<<(1024 * 1024 + 255) / 256, 256, 0, stream>>>(mi_w2, w_mi2, 1024, 1024);
    build_wcat_kernel<<<(NS * KZ + 255) / 256, 256, 0, stream>>>(W0, W1, W2, w_cat);

    // ---- fused pipeline per edge chunk ----
    for (int e0 = 0; e0 < NEDGE; e0 += CE) {
        edge_prep_kernel<<<CE / 8, 256, 0, stream>>>(nodes, pos, cell, edge_shift,
                                                     edge_index, batch_vec, ws_emb, ws_z, e0, CE);
        // h = silu(emb @ df_w1 + df_b1)
        gemm_bf16_kernel<1, 1, 1><<<dim3(CE / 128, 1024 / 128), 256, 0, stream>>>(
            ws_emb, w_df1, df_b1, ws_h, CE, 1024, 256);
        // distfilter = h @ df_w2 + df_b2                     (f32)
        gemm_bf16_kernel<0, 0, 1><<<dim3(CE / 128, 512 / 128), 256, 0, stream>>>(
            ws_h, w_df2, df_b2, ws_filt, CE, 512, 1024);
        // mixed = z @ Wcat                                   (f32)
        gemm_bf16_kernel<0, 0, 0><<<dim3(CE / 128, 512 / 128), 256, 0, stream>>>(
            ws_z, w_cat, nullptr, ws_mix, CE, 512, KZ);
        // reg = LN(mixed)*g+b * distfilter                   (bf16)
        ln_filter_kernel<<<CE / 8, 256, 0, stream>>>(ws_mix, ws_filt, ln_g, ln_b, ws_reg, CE);
        // h = silu(reg @ mi_w1 + mi_b1)
        gemm_bf16_kernel<1, 1, 1><<<dim3(CE / 128, 1024 / 128), 256, 0, stream>>>(
            ws_reg, w_mi1, mi_b1, ws_h, CE, 1024, 512);
        // h2 = silu(h @ mi_w2 + mi_b2)
        gemm_bf16_kernel<1, 1, 1><<<dim3(CE / 128, 1024 / 128), 256, 0, stream>>>(
            ws_h, w_mi2, mi_b2, ws_h2, CE, 1024, 1024);
        // out = h2 @ mo_w + mo_b
        out_kernel<<<CE / 8, 256, 0, stream>>>(ws_h2, mo_w, mo_b, out, e0);
    }
}
